// CubicalEcc_20624432956228
// MI455X (gfx1250) — compile-verified, matrix operand order checked by reading, not run on hardware
//
#include <hip/hip_runtime.h>

typedef __attribute__((ext_vector_type(16))) _Float16 v16h;
typedef __attribute__((ext_vector_type(8)))  float    v8f;

#define HH 224
#define WW 224
#define NSTEPS 32
#define NIMG 24                 // 8 * 3
#define HSTRIDE 257             // ints per bin row: 256 streams + 1 pad
#define NBINS 33                // bins 0..31 real, bin 32 = dead (v > t_max or +inf)
#define ROWS_PER_BLK 16
#define STRIPS 15               // ceil(225 / 16)

__device__ __forceinline__ int binof(float v) {
    // smallest k with v <= t_k, t_k = -2 + k*(4/31); 31/4 = 7.75 is exact in fp32
    float u = (v + 2.0f) * 7.75f;
    u = fminf(fmaxf(u, -1.0f), 33.0f);     // tames +inf before int convert
    int b = (int)ceilf(u);
    b = b < 0 ? 0 : b;
    b = b > 32 ? 32 : b;
    return b;
}

__global__ __launch_bounds__(256) void ecc_zero(float* __restrict__ out) {
    int i = blockIdx.x * blockDim.x + threadIdx.x;
    if (i < NIMG * NSTEPS) out[i] = 0.0f;
}

__global__ __launch_bounds__(256) void ecc_main(const float* __restrict__ x,
                                                float* __restrict__ out) {
    __shared__ int   hist[NBINS * HSTRIDE];
    __shared__ float Hs[2 * 16];            // reduced signed histogram, bins 0..31

    const int tid   = threadIdx.x;
    const int img   = blockIdx.x / STRIPS;  // 0..23 == b*3 + c
    const int strip = blockIdx.x % STRIPS;
    const int i0    = strip * ROWS_PER_BLK;
    const int rows  = min(ROWS_PER_BLK, (HH + 1) - i0);
    const float* __restrict__ im = x + (size_t)img * HH * WW;

    // zero this thread's private histogram column (bins 0..32)
    for (int b = 0; b < NBINS; ++b) hist[b * HSTRIDE + tid] = 0;
    __syncthreads();

    const float INF = __builtin_inff();
    const int npos = rows * (WW + 1);       // positions over (H+1) x (W+1) grid
    for (int p = tid; p < npos; p += 256) {
        int r = p / (WW + 1);
        int c = p - r * (WW + 1);
        int i = i0 + r;                     // 0..224
        int j = c;                          // 0..224
        float xc  = (i < HH && j < WW)     ? im[i * WW + j]             : INF;
        float xu  = (i > 0  && j < WW)     ? im[(i - 1) * WW + j]       : INF;
        float xl  = (i < HH && j > 0)      ? im[i * WW + (j - 1)]       : INF;
        float xul = (i > 0  && j > 0)      ? im[(i - 1) * WW + (j - 1)] : INF;

        float f  = xc;                      // face (2-cell)
        float eh = fminf(xu, xc);           // horizontal edge above pixel
        float ev = fminf(xl, xc);           // vertical edge left of pixel
        float vt = fminf(fminf(xul, xu), fminf(xl, xc));  // top-left vertex

        // signed Euler contributions: +F - Eh - Ev + V  (dead bin 32 absorbs inf)
        atomicAdd(&hist[binof(f)  * HSTRIDE + tid],  1);
        atomicAdd(&hist[binof(eh) * HSTRIDE + tid], -1);
        atomicAdd(&hist[binof(ev) * HSTRIDE + tid], -1);
        atomicAdd(&hist[binof(vt) * HSTRIDE + tid],  1);
    }
    __syncthreads();

    // Reduce 256 per-thread histograms over streams with WMMA:
    // D(16x16) += A(16 bins x 32 streams, f16) * B(ones 32x16).
    // B == all-ones makes the K-slot layout irrelevant; A row M = lane&15.
    const int wave = tid >> 5;
    const int lane = tid & 31;
    if (wave < 2) {
        const int tile = wave;                       // bins [tile*16, tile*16+16)
        const int bin  = tile * 16 + (lane & 15);
        v16h onesv;
#pragma unroll
        for (int k = 0; k < 16; ++k) onesv[k] = (_Float16)1.0f;

        v8f acc = {};
        for (int chunk = 0; chunk < 256; chunk += 32) {
            const int s0 = chunk + (lane & 16);      // lanes 0-15: +0, 16-31: +16
            v16h a;
#pragma unroll
            for (int k = 0; k < 16; ++k)
                a[k] = (_Float16)(float)hist[bin * HSTRIDE + s0 + k];
            acc = __builtin_amdgcn_wmma_f32_16x16x32_f16(
                false, a, false, onesv, (short)0, acc, false, false);
        }
        // D layout: lane L, VGPR r -> M = r + 8*(L>=16), N = L&15 (all N equal)
        if (lane == 0) {
#pragma unroll
            for (int r = 0; r < 8; ++r) Hs[tile * 16 + r] = acc[r];
        } else if (lane == 16) {
#pragma unroll
            for (int r = 0; r < 8; ++r) Hs[tile * 16 + 8 + r] = acc[r];
        }
    }
    __syncthreads();

    // cumulative sum over bins -> ECC partial for this strip; exact integer f32
    if (tid < NSTEPS) {
        float s = 0.0f;
        for (int b = 0; b <= tid; ++b) s += Hs[b];
        atomicAdd(&out[img * NSTEPS + tid], s);
    }
}

extern "C" void kernel_launch(void* const* d_in, const int* in_sizes, int n_in,
                              void* d_out, int out_size, void* d_ws, size_t ws_size,
                              hipStream_t stream) {
    (void)in_sizes; (void)n_in; (void)out_size; (void)d_ws; (void)ws_size;
    const float* x = (const float*)d_in[0];
    float* out = (float*)d_out;

    ecc_zero<<<dim3((NIMG * NSTEPS + 255) / 256), dim3(256), 0, stream>>>(out);
    ecc_main<<<dim3(NIMG * STRIPS), dim3(256), 0, stream>>>(x, out);
}
